// S_CLAM_29317446762504
// MI455X (gfx1250) — matrix operand order repaired and checked by reading, not compiled, not run on hardware
//
#include <hip/hip_runtime.h>
#include <stdint.h>

typedef __attribute__((ext_vector_type(16))) _Float16 v16h;
typedef __attribute__((ext_vector_type(8)))  _Float16 v8h;
typedef __attribute__((ext_vector_type(4)))  _Float16 v4h;
typedef __attribute__((ext_vector_type(8)))  float    v8f;
typedef __attribute__((ext_vector_type(4)))  float    v4f;

#define NBAG    100000
#define NFEAT   1024
#define NCOMP   512
#define NHID    256
#define MT      64
#define THREADS 512
#define NBLK    ((NBAG + MT - 1) / MT)   // 1563

// ---- d_out layout (float elements, concatenated tuple) ----
#define OFF_ATT  ((size_t)0)
#define OFF_A    ((size_t)NBAG * 2)
#define OFF_H    ((size_t)NBAG * 4)
#define OFF_LBL  (OFF_H + (size_t)NBAG * NCOMP)
#define OFF_INSU (OFF_LBL + 16)
#define OFF_INS  (OFF_INSU + 32)
#define OFF_SS   (OFF_INS + 32)
#define OFF_YP   (OFF_SS + 2)
#define OFF_YH   (OFF_YP + 2)

// ---- workspace layout (bytes) ----
#define WS_WCP   ((size_t)0)            // 1024x512 f16 packed = 1 MiB
#define WS_W12P  ((size_t)(1u << 20))   // 512x512 f16 packed  = 512 KiB
#define WS_PART  ((size_t)((1u << 20) + (1u << 19)))  // NBLK * 1024 f32 partials

// ---- LDS layout inside clam_main (bytes) ----
#define XS_STRIDE 264   // halves, K-chunk 256 + pad
#define HS_STRIDE 520   // halves, 512 + pad
#define AS_STRIDE 264   // halves, 256 + pad
#define LDS_XS    0
#define LDS_HS    (MT * XS_STRIDE * 2)                 // 33792
#define LDS_AS    (LDS_HS + MT * HS_STRIDE * 2)        // 100352
#define LDS_ATT   (LDS_AS + MT * AS_STRIDE * 2)        // 134144
#define LDS_PRB   (LDS_ATT + MT * 2 * 4)               // 134656
#define SMEM_BYTES (LDS_PRB + MT * 2 * 4)              // 135168

// =====================================================================
// Kernel 1: repack Wc (1024x512) and W1||W2 (512x512 combined) to f16 in
// WMMA B-fragment order: tile t = n_tile * NKT + k_tile, within a tile
// lane L holds 16 halves: W[k_tile*32 + (L>>4)*8 + (j<8 ? j : j+8),
// n_tile*16 + (L&15)], j = 0..15.  (ISA 7.12.2 / 7.12.4 B layout.)
// =====================================================================
__global__ void pack_weights(const float* __restrict__ Wc,
                             const float* __restrict__ W1,
                             const float* __restrict__ W2,
                             _Float16* __restrict__ WcP,
                             _Float16* __restrict__ W12P) {
  int tid = blockIdx.x * blockDim.x + threadIdx.x;
  const int WC_LANES  = (NFEAT / 32) * (NCOMP / 16) * 32;       // 32768
  const int W12_LANES = (NCOMP / 32) * ((2 * NHID) / 16) * 32;  // 16384
  if (tid < WC_LANES) {
    int lane = tid & 31, tile = tid >> 5;
    int nt = tile >> 5, kt = tile & 31;                 // NKT = 32
    int n = nt * 16 + (lane & 15);
    int kbase = kt * 32 + (lane >> 4) * 8;
    _Float16* dst = WcP + (size_t)tid * 16;
#pragma unroll
    for (int j = 0; j < 16; ++j)
      dst[j] = (_Float16)Wc[(size_t)(kbase + (j < 8 ? j : j + 8)) * NCOMP + n];
  } else if (tid < WC_LANES + W12_LANES) {
    int t2 = tid - WC_LANES;
    int lane = t2 & 31, tile = t2 >> 5;
    int nt = tile >> 4, kt = tile & 15;                 // NKT = 16
    int n = nt * 16 + (lane & 15);
    int kbase = kt * 32 + (lane >> 4) * 8;
    _Float16* dst = W12P + (size_t)t2 * 16;
#pragma unroll
    for (int j = 0; j < 16; ++j) {
      int k = kbase + (j < 8 ? j : j + 8);
      float v = (n < NHID) ? W1[(size_t)k * NHID + n]
                           : W2[(size_t)k * NHID + (n - NHID)];
      dst[j] = (_Float16)v;
    }
  }
}

// Build a v16h A-fragment from two LDS b128 loads (ISA 16-bit A layout:
// lanes 0-15 hold K {0..7,16..23}, lanes 16-31 hold K {8..15,24..31}).
static __device__ inline v16h load_afrag(const _Float16* p) {
  v8h lo = *(const v8h*)p;
  v8h hi = *(const v8h*)(p + 16);
  v16h a;
#pragma unroll
  for (int q = 0; q < 8; ++q) { a[q] = lo[q]; a[q + 8] = hi[q]; }
  return a;
}

// =====================================================================
// Kernel 2: fused main pipeline per 64-row tile:
//   h = relu(x@Wc+bc)  ->  a = tanh(h@W1+b1)*sigmoid(h@W2+b2)
//   att = a@W3+b3 -> softmax -> partial A^T @ h
// =====================================================================
__global__ __launch_bounds__(THREADS)
void clam_main(const float* __restrict__ x,
               const float* __restrict__ bc,
               const float* __restrict__ b1,
               const float* __restrict__ b2,
               const float* __restrict__ W3,
               const float* __restrict__ b3,
               const _Float16* __restrict__ WcP,
               const _Float16* __restrict__ W12P,
               float* __restrict__ out,
               float* __restrict__ partial) {
  extern __shared__ char smem[];
  _Float16* xs   = (_Float16*)(smem + LDS_XS);
  _Float16* hs   = (_Float16*)(smem + LDS_HS);
  _Float16* as_  = (_Float16*)(smem + LDS_AS);
  float*    atts = (float*)(smem + LDS_ATT);
  float*    prbs = (float*)(smem + LDS_PRB);

  const int tid  = threadIdx.x;
  const int lane = tid & 31;
  const int w    = tid >> 5;           // 16 waves
  const int m0   = blockIdx.x * MT;
  const int lrow = lane & 15;
  const int hi8  = (lane >> 4) * 8;    // lane>=16 handles M/K upper half

  // -------- Phase 1: h = relu(x @ Wc + bc), K streamed in 256-chunks ----
  v8f acc[4][2];
#pragma unroll
  for (int ms = 0; ms < 4; ++ms)
#pragma unroll
    for (int j = 0; j < 2; ++j) acc[ms][j] = (v8f){};

  const int nt0 = 2 * w;  // phase-1 N-tiles {2w, 2w+1}
  for (int kc = 0; kc < NFEAT / 256; ++kc) {
    // cooperative load + f32->f16 convert of x[m0:m0+64, kc*256:+256]
#pragma unroll
    for (int it = 0; it < (MT * 256 / 4) / THREADS; ++it) {  // 8 float4 each
      int linear = tid + it * THREADS;
      int r  = linear >> 6;        // 64 float4 per row
      int c4 = linear & 63;
      int rg = m0 + r;
      v4f f = (v4f){};
      if (rg < NBAG)
        f = *(const v4f*)(x + (size_t)rg * NFEAT + kc * 256 + c4 * 4);
      v4h h4;
      h4[0] = (_Float16)f[0]; h4[1] = (_Float16)f[1];
      h4[2] = (_Float16)f[2]; h4[3] = (_Float16)f[3];
      *(v4h*)(xs + r * XS_STRIDE + c4 * 4) = h4;
    }
    __syncthreads();

    for (int kfl = 0; kfl < 8; ++kfl) {
      int kf = kc * 8 + kfl;
      v16h B0 = *(const v16h*)(WcP + ((size_t)((nt0 + 0) * 32 + kf) * 32 + lane) * 16);
      v16h B1 = *(const v16h*)(WcP + ((size_t)((nt0 + 1) * 32 + kf) * 32 + lane) * 16);
#pragma unroll
      for (int ms = 0; ms < 4; ++ms) {
        v16h A = load_afrag(xs + (ms * 16 + lrow) * XS_STRIDE + kfl * 32 + hi8);
        acc[ms][0] = __builtin_amdgcn_wmma_f32_16x16x32_f16(
            false, A, false, B0, (short)0, acc[ms][0], false, false);
        acc[ms][1] = __builtin_amdgcn_wmma_f32_16x16x32_f16(
            false, A, false, B1, (short)0, acc[ms][1], false, false);
      }
    }
    __syncthreads();
  }

  // epilogue: +bc, relu, store h (f32 global + f16 LDS)
#pragma unroll
  for (int j = 0; j < 2; ++j) {
    int col = (nt0 + j) * 16 + lrow;
    float bcv = bc[col];
#pragma unroll
    for (int ms = 0; ms < 4; ++ms)
#pragma unroll
      for (int i = 0; i < 8; ++i) {
        float v = acc[ms][j][i] + bcv;
        v = fmaxf(v, 0.0f);
        int rl = ms * 16 + hi8 + i;
        hs[rl * HS_STRIDE + col] = (_Float16)v;
        int rg = m0 + rl;
        if (rg < NBAG) out[OFF_H + (size_t)rg * NCOMP + col] = v;
      }
  }
  __syncthreads();

  // -------- Phase 2: gated attention, wave w owns column pair (n, n+256) --
  v8f acc2[4][2];
#pragma unroll
  for (int ms = 0; ms < 4; ++ms)
#pragma unroll
    for (int j = 0; j < 2; ++j) acc2[ms][j] = (v8f){};

  const int ntA = w;        // W1 side (cols w*16..)
  const int ntB = w + 16;   // W2 side (cols 256 + w*16..)
  for (int kf = 0; kf < NCOMP / 32; ++kf) {
    v16h B0 = *(const v16h*)(W12P + ((size_t)(ntA * 16 + kf) * 32 + lane) * 16);
    v16h B1 = *(const v16h*)(W12P + ((size_t)(ntB * 16 + kf) * 32 + lane) * 16);
#pragma unroll
    for (int ms = 0; ms < 4; ++ms) {
      v16h A = load_afrag(hs + (ms * 16 + lrow) * HS_STRIDE + kf * 32 + hi8);
      acc2[ms][0] = __builtin_amdgcn_wmma_f32_16x16x32_f16(
          false, A, false, B0, (short)0, acc2[ms][0], false, false);
      acc2[ms][1] = __builtin_amdgcn_wmma_f32_16x16x32_f16(
          false, A, false, B1, (short)0, acc2[ms][1], false, false);
    }
  }
  {
    int col = w * 16 + lrow;               // 0..255
    float b1v = b1[col], b2v = b2[col];
#pragma unroll
    for (int ms = 0; ms < 4; ++ms)
#pragma unroll
      for (int i = 0; i < 8; ++i) {
        float u1 = acc2[ms][0][i] + b1v;
        float u2 = acc2[ms][1][i] + b2v;
        float a  = tanhf(u1) * (1.0f / (1.0f + expf(-u2)));
        int rl = ms * 16 + hi8 + i;
        as_[rl * AS_STRIDE + col] = (_Float16)a;
      }
  }
  __syncthreads();

  // -------- Phase 3: att = a @ W3 + b3 (N=2), softmax -------------------
  if (tid < MT * 2) {
    int r = tid >> 1, c = tid & 1;
    float s = b3[c];
    for (int jj = 0; jj < NHID; ++jj)
      s += (float)as_[r * AS_STRIDE + jj] * W3[jj * 2 + c];
    atts[r * 2 + c] = s;
    int rg = m0 + r;
    if (rg < NBAG) out[OFF_ATT + (size_t)rg * 2 + c] = s;
  }
  __syncthreads();
  if (tid < MT) {
    int r = tid, rg = m0 + r;
    float s0 = atts[2 * r], s1 = atts[2 * r + 1];
    float mx = fmaxf(s0, s1);
    float e0 = expf(s0 - mx), e1 = expf(s1 - mx);
    float inv = 1.0f / (e0 + e1);
    float p0 = e0 * inv, p1 = e1 * inv;
    if (rg < NBAG) {
      out[OFF_A + (size_t)rg * 2 + 0] = p0;
      out[OFF_A + (size_t)rg * 2 + 1] = p1;
      prbs[2 * r] = p0; prbs[2 * r + 1] = p1;
    } else {
      prbs[2 * r] = 0.0f; prbs[2 * r + 1] = 0.0f;  // kill padded rows
    }
  }
  __syncthreads();

  // -------- Phase 4: partial slide_agg = A^T @ h (2 x 512) --------------
  {
    int d = tid;  // THREADS == NCOMP
    float a0 = 0.0f, a1 = 0.0f;
    for (int r = 0; r < MT; ++r) {
      float hv = (float)hs[r * HS_STRIDE + d];
      a0 += prbs[2 * r] * hv;
      a1 += prbs[2 * r + 1] * hv;
    }
    partial[(size_t)blockIdx.x * 1024 + d]       = a0;
    partial[(size_t)blockIdx.x * 1024 + 512 + d] = a1;
  }
}

// monotonic float->uint key
static __device__ inline unsigned fkey(float v) {
  unsigned u = __float_as_uint(v);
  return (u & 0x80000000u) ? ~u : (u | 0x80000000u);
}

// =====================================================================
// Kernel 3: global top-8 / bottom-8 of A[:, slide_label] + instance
// classifier on the 16 gathered h rows.
// =====================================================================
__global__ __launch_bounds__(1024)
void topk_ins(const float* __restrict__ ro,   // d_out (read A, h)
              float* __restrict__ out,
              const int* __restrict__ slide_label,
              const float* __restrict__ W_ins,
              const float* __restrict__ b_ins) {
  __shared__ unsigned long long rk[1024];
  __shared__ int sel[16];
  const int tid = threadIdx.x;
  const int sl = slide_label[0];

  float tv[8]; int ti[8]; float bv[8]; int bi[8];
#pragma unroll
  for (int p = 0; p < 8; ++p) {
    tv[p] = -3.0e38f; ti[p] = -1; bv[p] = 3.0e38f; bi[p] = -1;
  }
  for (int i = tid; i < NBAG; i += 1024) {
    float v = ro[OFF_A + (size_t)i * 2 + sl];
    if (v > tv[7]) {
      int p = 7;
      while (p > 0 && tv[p - 1] < v) { tv[p] = tv[p - 1]; ti[p] = ti[p - 1]; --p; }
      tv[p] = v; ti[p] = i;
    }
    if (v < bv[7]) {
      int p = 7;
      while (p > 0 && bv[p - 1] > v) { bv[p] = bv[p - 1]; bi[p] = bi[p - 1]; --p; }
      bv[p] = v; bi[p] = i;
    }
  }

  // 8 rounds of packed-key max-reduction for the top set
  for (int s = 0; s < 8; ++s) {
    unsigned long long best = 0ull;
#pragma unroll
    for (int p = 0; p < 8; ++p)
      if (ti[p] >= 0) {
        unsigned long long k =
            ((unsigned long long)fkey(tv[p]) << 32) |
            (unsigned long long)(0xFFFFFFFFu - (unsigned)ti[p]);
        if (k > best) best = k;
      }
    rk[tid] = best;
    __syncthreads();
    for (int off = 512; off > 0; off >>= 1) {
      if (tid < off && rk[tid + off] > rk[tid]) rk[tid] = rk[tid + off];
      __syncthreads();
    }
    unsigned long long win = rk[0];
    __syncthreads();
    int idx = (int)(0xFFFFFFFFu - (unsigned)(win & 0xFFFFFFFFu));
    if (tid == 0) sel[s] = idx;
#pragma unroll
    for (int p = 0; p < 8; ++p)
      if (ti[p] == idx) ti[p] = -1;
  }
  // 8 rounds for the bottom set (max over key of -v)
  for (int s = 0; s < 8; ++s) {
    unsigned long long best = 0ull;
#pragma unroll
    for (int p = 0; p < 8; ++p)
      if (bi[p] >= 0) {
        unsigned long long k =
            ((unsigned long long)fkey(-bv[p]) << 32) |
            (unsigned long long)(0xFFFFFFFFu - (unsigned)bi[p]);
        if (k > best) best = k;
      }
    rk[tid] = best;
    __syncthreads();
    for (int off = 512; off > 0; off >>= 1) {
      if (tid < off && rk[tid + off] > rk[tid]) rk[tid] = rk[tid + off];
      __syncthreads();
    }
    unsigned long long win = rk[0];
    __syncthreads();
    int idx = (int)(0xFFFFFFFFu - (unsigned)(win & 0xFFFFFFFFu));
    if (tid == 0) sel[8 + s] = idx;
#pragma unroll
    for (int p = 0; p < 8; ++p)
      if (bi[p] == idx) bi[p] = -1;
  }
  __syncthreads();

  if (tid < 16) {
    int row = sel[tid];
    float l0 = b_ins[0], l1 = b_ins[1];
    for (int d = 0; d < NCOMP; ++d) {
      float hv = ro[OFF_H + (size_t)row * NCOMP + d];
      l0 += hv * W_ins[d * 2 + 0];
      l1 += hv * W_ins[d * 2 + 1];
    }
    out[OFF_INSU + tid * 2 + 0] = l0;
    out[OFF_INSU + tid * 2 + 1] = l1;
    float mx = fmaxf(l0, l1);
    float e0 = expf(l0 - mx), e1 = expf(l1 - mx);
    float inv = 1.0f / (e0 + e1);
    out[OFF_INS + tid * 2 + 0] = e0 * inv;
    out[OFF_INS + tid * 2 + 1] = e1 * inv;
    out[OFF_LBL + tid] = (tid < 8) ? 1.0f : 0.0f;
  }
}

// =====================================================================
// Kernel 4: reduce slide_agg partials, slide_agg @ W_bag + b_bag,
// softmax + argmax.
// =====================================================================
__global__ __launch_bounds__(1024)
void finalize(const float* __restrict__ partial, int nblk,
              const float* __restrict__ W_bag,
              const float* __restrict__ b_bag,
              float* __restrict__ out) {
  __shared__ float agg[1024];
  __shared__ float sc[2];
  const int tid = threadIdx.x;
  float s = 0.0f;
  for (int b = 0; b < nblk; ++b) s += partial[(size_t)b * 1024 + tid];
  agg[tid] = s;
  __syncthreads();
  if (tid < 2) {
    float v = b_bag[0];
    for (int d = 0; d < NCOMP; ++d) v += agg[tid * NCOMP + d] * W_bag[d];
    sc[tid] = v;
    out[OFF_SS + tid] = v;
  }
  __syncthreads();
  if (tid == 0) {
    float s0 = sc[0], s1 = sc[1];
    float mx = fmaxf(s0, s1);
    float e0 = expf(s0 - mx), e1 = expf(s1 - mx);
    float inv = 1.0f / (e0 + e1);
    out[OFF_YP + 0] = e0 * inv;
    out[OFF_YP + 1] = e1 * inv;
    out[OFF_YH] = (s1 > s0) ? 1.0f : 0.0f;
  }
}

extern "C" void kernel_launch(void* const* d_in, const int* in_sizes, int n_in,
                              void* d_out, int out_size, void* d_ws, size_t ws_size,
                              hipStream_t stream) {
  (void)in_sizes; (void)n_in; (void)out_size; (void)ws_size;
  const float* x     = (const float*)d_in[0];
  const int*   slide = (const int*)d_in[1];
  const float* Wc    = (const float*)d_in[2];
  const float* bc    = (const float*)d_in[3];
  const float* W1    = (const float*)d_in[4];
  const float* b1    = (const float*)d_in[5];
  const float* W2    = (const float*)d_in[6];
  const float* b2    = (const float*)d_in[7];
  const float* W3    = (const float*)d_in[8];
  const float* b3    = (const float*)d_in[9];
  const float* Wins  = (const float*)d_in[10];
  const float* bins  = (const float*)d_in[11];
  const float* Wbag  = (const float*)d_in[12];
  const float* bbag  = (const float*)d_in[13];
  float* out = (float*)d_out;
  char*  ws  = (char*)d_ws;

  _Float16* WcP  = (_Float16*)(ws + WS_WCP);
  _Float16* W12P = (_Float16*)(ws + WS_W12P);
  float* partial = (float*)(ws + WS_PART);

  pack_weights<<<192, 256, 0, stream>>>(Wc, W1, W2, WcP, W12P);
  clam_main<<<NBLK, THREADS, SMEM_BYTES, stream>>>(
      x, bc, b1, b2, W3, b3, WcP, W12P, out, partial);
  topk_ins<<<1, 1024, 0, stream>>>(out, out, slide, Wins, bins);
  finalize<<<1, 1024, 0, stream>>>(partial, NBLK, Wbag, bbag, out);
}
